// LearnedPrimalDual_16707422781591
// MI455X (gfx1250) — compile-verified, matrix operand order checked by reading, not run on hardware
//
#include <hip/hip_runtime.h>
#include <math.h>

typedef unsigned short ushort_t;
typedef __bf16 bf16;
typedef bf16 v16bf __attribute__((ext_vector_type(16)));
typedef float v8f  __attribute__((ext_vector_type(8)));

static constexpr int   Bn = 4, Hn = 256, Wn = 256, An = 180, Dn = 256, Sn = 256;
static constexpr float OPN_INV = 1.0f / 215.0f;
static constexpr float DEG     = 3.14159265358979323846f / 180.0f;

__device__ __forceinline__ ushort_t f2bf(float f) {
  union { float f; unsigned int u; } x; x.f = f;
  unsigned int r = x.u + 0x7FFFu + ((x.u >> 16) & 1u);
  return (ushort_t)(r >> 16);
}

union FragU { v16bf v; uint4 q[2]; ushort_t u[16]; };

// ---------------------------------------------------------------------------
// Weight packing: HWIO f32 -> bf16 B-fragment layout [chunk][ntile][lane][16]
// K index mapping (16-bit operand, 16x16x32):
//   klocal = (e/8)*16 + (lane/16)*8 + (e%8);  kglobal = chunk*32 + klocal
//   kglobal -> tap = k / CIN_PAD, ci = k % CIN_PAD ; N = ntile*16 + lane%16
// ---------------------------------------------------------------------------
__global__ void lpd_pack_weights(const float* __restrict__ w, ushort_t* __restrict__ dst,
                                 int cin, int cout, int cin_pad, int nchunk, int ntiles) {
  int idx = blockIdx.x * blockDim.x + threadIdx.x;
  int total = nchunk * ntiles * 512;
  if (idx >= total) return;
  int e    = idx & 15;
  int lane = (idx >> 4) & 31;
  int t    = (idx >> 9) % ntiles;
  int c    = (idx >> 9) / ntiles;
  int klocal = (e >> 3) * 16 + (lane >> 4) * 8 + (e & 7);
  int kg  = c * 32 + klocal;
  int tap = kg / cin_pad;
  int ci  = kg % cin_pad;
  int co  = t * 16 + (lane & 15);
  float v = 0.f;
  if (tap < 9 && ci < cin && co < cout) v = w[(tap * cin + ci) * cout + co];
  dst[idx] = f2bf(v);
}

// ---------------------------------------------------------------------------
// Implicit-GEMM 3x3 SAME conv via v_wmma_f32_16x16x32_bf16.
// Block: 128 threads = 4 wave32; tile = 64 pixels (along W) x cout.
// LDS: 3 rows x 66 cols x CIN_PAD bf16 input patch, zero border,
//      plus one zeroed 16B slot used as the target for padded K-taps
//      (offset select -> ds_load_b128, never a flat/scratch select).
// ---------------------------------------------------------------------------
template <int CIN_PAD, int NCHUNK, int NTILES, bool RELU_OUT>
__global__ __launch_bounds__(128) void lpd_conv3x3_wmma(
    const ushort_t* __restrict__ X,   // [B,Hc,Wc,CIN_PAD] bf16 bits
    const ushort_t* __restrict__ Wp,  // packed weights
    const float*    __restrict__ bias,
    ushort_t*       __restrict__ Hout, // RELU: [B,Hc,Wc,32] bf16
    float*          __restrict__ Racc, // !RELU: [B,Hc,Wc,5] f32 (+=)
    int Hc, int Wc) {
  constexpr int LW   = 66;
  constexpr int ZOFF = 3 * LW * CIN_PAD;  // zero slot (ushort index, 16B aligned)
  __shared__ alignas(16) ushort_t smem[ZOFF + 8];
  const int tx   = threadIdx.x;
  const int lane = tx & 31;
  const int wave = tx >> 5;
  const int x0   = blockIdx.x * 64;
  const int yr   = blockIdx.y;
  const int b    = blockIdx.z;

  constexpr int CP8 = CIN_PAD / 8;
  for (int i = tx; i < 3 * LW * CP8; i += 128) {
    int sub = i % CP8;
    int pix = i / CP8;
    int col = pix % LW;
    int row = pix / LW;
    int gy = yr + row - 1;
    int gx = x0 + col - 1;
    uint4 v = make_uint4(0, 0, 0, 0);
    if (gy >= 0 && gy < Hc && gx >= 0 && gx < Wc)
      v = *(const uint4*)(X + (((size_t)(b * Hc + gy) * Wc + gx) * CIN_PAD + sub * 8));
    *(uint4*)(smem + (size_t)pix * CIN_PAD + sub * 8) = v;
  }
  if (tx == 0) *(uint4*)(smem + ZOFF) = make_uint4(0, 0, 0, 0);
  __syncthreads();

  const int khalf = lane >> 4;               // which K-half this lane holds
  const int pxl   = wave * 16 + (lane & 15); // local pixel (A-matrix row M)
  v8f zero8 = {0.f, 0.f, 0.f, 0.f, 0.f, 0.f, 0.f, 0.f};
  v8f acc[NTILES];
#pragma unroll
  for (int t = 0; t < NTILES; ++t) acc[t] = zero8;

#pragma unroll
  for (int c = 0; c < NCHUNK; ++c) {
    FragU a;
    if constexpr (CIN_PAD == 32) {
      const int dy = c / 3, dx = c - dy * 3;  // one tap per K-chunk
      const ushort_t* base = smem + ((size_t)(dy * LW) + pxl + dx) * 32 + khalf * 8;
      a.q[0] = *(const uint4*)(base);
      a.q[1] = *(const uint4*)(base + 16);
    } else {  // CIN_PAD == 8: two taps per K-chunk half; padded taps -> zero slot
      const int tapA = c * 4 + khalf;
      const int tapB = tapA + 2;
      const int offA = (tapA < 9) ? (((tapA / 3) * LW + pxl + (tapA % 3)) * 8) : ZOFF;
      const int offB = (tapB < 9) ? (((tapB / 3) * LW + pxl + (tapB % 3)) * 8) : ZOFF;
      a.q[0] = *(const uint4*)(smem + offA);
      a.q[1] = *(const uint4*)(smem + offB);
    }
#pragma unroll
    for (int t = 0; t < NTILES; ++t) {
      FragU bf;
      const uint4* bp = (const uint4*)(Wp + (((size_t)c * NTILES + t) * 32 + lane) * 16);
      bf.q[0] = bp[0];
      bf.q[1] = bp[1];
      acc[t] = __builtin_amdgcn_wmma_f32_16x16x32_bf16(false, a.v, false, bf.v,
                                                       (short)0, acc[t], false, false);
    }
  }

  const int n = lane & 15;  // D-matrix column
  if constexpr (RELU_OUT) {
#pragma unroll
    for (int t = 0; t < NTILES; ++t) {
      const int co = t * 16 + n;
      const float bb = bias[co];
#pragma unroll
      for (int r = 0; r < 8; ++r) {
        int px = x0 + wave * 16 + r + khalf * 8;  // D row M = r + (lane/16)*8
        float v = acc[t][r] + bb;
        v = v > 0.f ? v : 0.f;
        Hout[((size_t)(b * Hc + yr) * Wc + px) * 32 + co] = f2bf(v);
      }
    }
  } else {
    if (n < 5) {
      const float bb = bias[n];
#pragma unroll
      for (int r = 0; r < 8; ++r) {
        int px = x0 + wave * 16 + r + khalf * 8;
        size_t o = ((size_t)(b * Hc + yr) * Wc + px) * 5 + n;
        Racc[o] += acc[0][r] + bb;  // residual update of primal/dual
      }
    }
  }
}

// ---------------------------------------------------------------------------
// Radon forward: p(b,a,d) = sum_s bilinear(primal[...,1], x,y) / 215
// ---------------------------------------------------------------------------
__global__ void lpd_radon_fwd(const float* __restrict__ primal, float* __restrict__ evalop) {
  int i = blockIdx.x * blockDim.x + threadIdx.x;
  if (i >= Bn * An * Dn) return;
  int d = i % Dn;
  int a = (i / Dn) % An;
  int b = i / (Dn * An);
  float th = a * DEG;
  float ct = cosf(th), st = sinf(th);
  float u = d - 127.5f;
  const float* img = primal + (size_t)b * Hn * Wn * 5 + 1;  // channel 1, stride 5
  float acc = 0.f;
  for (int s = 0; s < Sn; ++s) {
    float t = s - 127.5f;
    float x = 127.5f + u * ct - t * st;
    float y = 127.5f + u * st + t * ct;
    float xf = floorf(x), yf = floorf(y);
    int xi = (int)xf, yi = (int)yf;
    float fx = x - xf, fy = y - yf;
    if ((unsigned)yi < (unsigned)Hn && (unsigned)xi < (unsigned)Wn)
      acc += (1.f - fx) * (1.f - fy) * img[((size_t)yi * Wn + xi) * 5];
    if ((unsigned)yi < (unsigned)Hn && (unsigned)(xi + 1) < (unsigned)Wn)
      acc += fx * (1.f - fy) * img[((size_t)yi * Wn + xi + 1) * 5];
    if ((unsigned)(yi + 1) < (unsigned)Hn && (unsigned)xi < (unsigned)Wn)
      acc += (1.f - fx) * fy * img[((size_t)(yi + 1) * Wn + xi) * 5];
    if ((unsigned)(yi + 1) < (unsigned)Hn && (unsigned)(xi + 1) < (unsigned)Wn)
      acc += fx * fy * img[((size_t)(yi + 1) * Wn + xi + 1) * 5];
  }
  evalop[i] = acc * OPN_INV;
}

// ---------------------------------------------------------------------------
// Exact adjoint as a deterministic gather (no atomics): bilinear splat weights
// are tensor-product hats, so pixel p receives hat(x-px)*hat(y-py) from every
// sample; candidates lie in a 3x3 (d,s) window around the rotated pixel coord.
// ---------------------------------------------------------------------------
__global__ void lpd_radon_adj(const float* __restrict__ dual, float* __restrict__ adj) {
  __shared__ float sct[An], sst[An];
  int tid = threadIdx.x;
  for (int a = tid; a < An; a += blockDim.x) {
    float th = a * DEG;
    sct[a] = cosf(th);
    sst[a] = sinf(th);
  }
  __syncthreads();
  int i = blockIdx.x * blockDim.x + tid;
  if (i >= Bn * Hn * Wn) return;
  int px = i % Wn;
  int py = (i / Wn) % Hn;
  int b  = i / (Hn * Wn);
  float dx = px - 127.5f, dy = py - 127.5f;
  const float* sino = dual + (size_t)b * An * Dn * 5;  // channel 0, stride 5
  float acc = 0.f;
  for (int a = 0; a < An; ++a) {
    float ct = sct[a], st = sst[a];
    float dc = dx * ct + dy * st + 127.5f;   // detector coord of pixel
    float sc = -dx * st + dy * ct + 127.5f;  // sample coord of pixel
    int d0 = (int)ceilf(dc - 1.4143f);
    int s0 = (int)ceilf(sc - 1.4143f);
    for (int jd = 0; jd < 3; ++jd) {
      int d = d0 + jd;
      if ((unsigned)d >= (unsigned)Dn) continue;
      float u = d - 127.5f;
      float bx = u * ct, by = u * st;
      float sval = sino[((size_t)a * Dn + d) * 5];
      for (int js = 0; js < 3; ++js) {
        int s = s0 + js;
        if ((unsigned)s >= (unsigned)Sn) continue;
        float t = s - 127.5f;
        float wx = 1.f - fabsf(bx - t * st - dx);
        if (wx <= 0.f) continue;
        float wy = 1.f - fabsf(by + t * ct - dy);
        if (wy <= 0.f) continue;
        acc += wx * wy * sval;
      }
    }
  }
  adj[i] = acc * OPN_INV;
}

// ---------------------------------------------------------------------------
// Input packers (f32 -> channel-padded bf16), fill, output copy
// ---------------------------------------------------------------------------
__global__ void lpd_pack_dual_in(const float* __restrict__ dual, const float* __restrict__ evalop,
                                 const float* __restrict__ y, ushort_t* __restrict__ Xd) {
  int i = blockIdx.x * blockDim.x + threadIdx.x;
  if (i >= Bn * An * Dn) return;
  size_t o = (size_t)i * 8;
#pragma unroll
  for (int c = 0; c < 5; ++c) Xd[o + c] = f2bf(dual[(size_t)i * 5 + c]);
  Xd[o + 5] = f2bf(evalop[i]);
  Xd[o + 6] = f2bf(y[i]);
  Xd[o + 7] = 0;
}

__global__ void lpd_pack_primal_in(const float* __restrict__ primal, const float* __restrict__ adj,
                                   ushort_t* __restrict__ Xp) {
  int i = blockIdx.x * blockDim.x + threadIdx.x;
  if (i >= Bn * Hn * Wn) return;
  size_t o = (size_t)i * 8;
#pragma unroll
  for (int c = 0; c < 5; ++c) Xp[o + c] = f2bf(primal[(size_t)i * 5 + c]);
  Xp[o + 5] = f2bf(adj[i]);
  Xp[o + 6] = 0;
  Xp[o + 7] = 0;
}

__global__ void lpd_fill_zero(float* __restrict__ p, int n) {
  int i = blockIdx.x * blockDim.x + threadIdx.x;
  if (i < n) p[i] = 0.f;
}

__global__ void lpd_copy_out(const float* __restrict__ primal, float* __restrict__ out) {
  int i = blockIdx.x * blockDim.x + threadIdx.x;
  if (i < Bn * Hn * Wn) out[i] = primal[(size_t)i * 5];
}

// ---------------------------------------------------------------------------
extern "C" void kernel_launch(void* const* d_in, const int* in_sizes, int n_in,
                              void* d_out, int out_size, void* d_ws, size_t ws_size,
                              hipStream_t stream) {
  if (n_in < 1 + 10 * 12) return;
  const float* yin = (const float*)d_in[0];
  auto win = [&](int it, int j) { return (const float*)d_in[1 + it * 12 + 2 * j]; };
  auto bin = [&](int it, int j) { return (const float*)d_in[1 + it * 12 + 2 * j + 1]; };

  char* ws = (char*)d_ws;
  size_t off = 0;
  auto alloc = [&](size_t bytes) -> void* {
    void* p = ws + off;
    off = (off + bytes + 255) & ~(size_t)255;
    return p;
  };
  float*    primal = (float*)alloc((size_t)Bn * Hn * Wn * 5 * 4);
  float*    dual   = (float*)alloc((size_t)Bn * An * Dn * 5 * 4);
  float*    evalop = (float*)alloc((size_t)Bn * An * Dn * 4);
  float*    adj    = (float*)alloc((size_t)Bn * Hn * Wn * 4);
  ushort_t* Xd     = (ushort_t*)alloc((size_t)Bn * An * Dn * 8 * 2);
  ushort_t* Xp     = (ushort_t*)alloc((size_t)Bn * Hn * Wn * 8 * 2);
  ushort_t* H1     = (ushort_t*)alloc((size_t)Bn * Hn * Wn * 32 * 2);
  ushort_t* H2     = (ushort_t*)alloc((size_t)Bn * Hn * Wn * 32 * 2);
  static const int cin_[6]    = {7, 32, 32, 6, 32, 32};
  static const int cinpad_[6] = {8, 32, 32, 8, 32, 32};
  static const int nch_[6]    = {3, 9, 9, 3, 9, 9};
  static const int nti_[6]    = {2, 2, 1, 2, 2, 1};
  static const int cout_[6]   = {32, 32, 5, 32, 32, 5};
  size_t wof[6], wstride = 0;
  for (int j = 0; j < 6; ++j) { wof[j] = wstride; wstride += (size_t)nch_[j] * nti_[j] * 512; }
  ushort_t* wpack = (ushort_t*)alloc(wstride * 10 * 2);
  if (off > ws_size) return;

  // one-time (per launch) weight packing into WMMA B-fragment layout
  for (int it = 0; it < 10; ++it)
    for (int j = 0; j < 6; ++j) {
      int total = nch_[j] * nti_[j] * 512;
      lpd_pack_weights<<<(total + 255) / 256, 256, 0, stream>>>(
          win(it, j), wpack + wstride * it + wof[j],
          cin_[j], cout_[j], cinpad_[j], nch_[j], nti_[j]);
    }

  lpd_fill_zero<<<(Bn * Hn * Wn * 5 + 255) / 256, 256, 0, stream>>>(primal, Bn * Hn * Wn * 5);
  lpd_fill_zero<<<(Bn * An * Dn * 5 + 255) / 256, 256, 0, stream>>>(dual, Bn * An * Dn * 5);

  const int nSino = Bn * An * Dn;   // 184320
  const int nImg  = Bn * Hn * Wn;   // 262144

  for (int it = 0; it < 10; ++it) {
    ushort_t* wpi = wpack + wstride * it;
    // ---- dual half: dual += block([dual, R(primal[:,1]), y]) on 180x256 grid
    lpd_radon_fwd<<<(nSino + 255) / 256, 256, 0, stream>>>(primal, evalop);
    lpd_pack_dual_in<<<(nSino + 255) / 256, 256, 0, stream>>>(dual, evalop, yin, Xd);
    lpd_conv3x3_wmma<8, 3, 2, true><<<dim3(Dn / 64, An, Bn), 128, 0, stream>>>(
        Xd, wpi + wof[0], bin(it, 0), H1, nullptr, An, Dn);
    lpd_conv3x3_wmma<32, 9, 2, true><<<dim3(Dn / 64, An, Bn), 128, 0, stream>>>(
        H1, wpi + wof[1], bin(it, 1), H2, nullptr, An, Dn);
    lpd_conv3x3_wmma<32, 9, 1, false><<<dim3(Dn / 64, An, Bn), 128, 0, stream>>>(
        H2, wpi + wof[2], bin(it, 2), nullptr, dual, An, Dn);
    // ---- primal half: primal += block([primal, R^T(dual[:,0])]) on 256x256
    lpd_radon_adj<<<(nImg + 255) / 256, 256, 0, stream>>>(dual, adj);
    lpd_pack_primal_in<<<(nImg + 255) / 256, 256, 0, stream>>>(primal, adj, Xp);
    lpd_conv3x3_wmma<8, 3, 2, true><<<dim3(Wn / 64, Hn, Bn), 128, 0, stream>>>(
        Xp, wpi + wof[3], bin(it, 3), H1, nullptr, Hn, Wn);
    lpd_conv3x3_wmma<32, 9, 2, true><<<dim3(Wn / 64, Hn, Bn), 128, 0, stream>>>(
        H1, wpi + wof[4], bin(it, 4), H2, nullptr, Hn, Wn);
    lpd_conv3x3_wmma<32, 9, 1, false><<<dim3(Wn / 64, Hn, Bn), 128, 0, stream>>>(
        H2, wpi + wof[5], bin(it, 5), nullptr, primal, Hn, Wn);
  }

  lpd_copy_out<<<(nImg + 255) / 256, 256, 0, stream>>>(primal, (float*)d_out);
}
